// LinearLSTMModel_40913858461766
// MI455X (gfx1250) — matrix operand order repaired and checked by reading, not compiled, not run on hardware
//
#include <hip/hip_runtime.h>

// ---------------------------------------------------------------------------
// Types: clang ext_vector types (trivial, union-safe, guarantee b128 ops)
// ---------------------------------------------------------------------------
typedef __bf16    v16bf  __attribute__((ext_vector_type(16)));
typedef float     v8f    __attribute__((ext_vector_type(8)));
typedef unsigned  uint4e __attribute__((ext_vector_type(4)));
typedef float     flt4e  __attribute__((ext_vector_type(4)));

union FragBF { v16bf v; uint4e q[2]; };

__device__ __forceinline__ float sigm(float x) { return 1.f / (1.f + __expf(-x)); }
__device__ __forceinline__ float bflo(unsigned u) { return __uint_as_float(u << 16); }
__device__ __forceinline__ float bfhi(unsigned u) { return __uint_as_float(u & 0xffff0000u); }

// A fragment (16x32 bf16, MxK), row-major activations, per documented layout:
// lanes 0-15: M=lane, elems0-7 = K k0..k0+7, elems8-15 = K k0+16..k0+23
// lanes16-31: M=lane-16, elems0-7 = K k0+8..k0+15, elems8-15 = K k0+24..k0+31
__device__ __forceinline__ v16bf load_a(const __bf16* A, int lda, int m, int k0, int lane) {
  FragBF f;
  const char* p = (const char*)(A + (size_t)m * lda + (k0 + ((lane >> 4) << 3)));
  f.q[0] = *(const uint4e*)p;
  f.q[1] = *(const uint4e*)(p + 32);
  return f.v;
}

// B fragment (32x16 bf16, KxN) where B[k][n] = W[n][k], W row-major [Nout,K]:
// lanes 0-15: N=lane, elems0-7 = K k0..k0+7, elems8-15 = K k0+16..k0+23
// lanes16-31: N=lane-16, elems0-7 = K k0+8..k0+15, elems8-15 = K k0+24..k0+31
__device__ __forceinline__ v16bf load_b(const __bf16* W, int ldw, int n, int k0, int lane) {
  FragBF f;
  const char* p = (const char*)(W + (size_t)n * ldw + (k0 + ((lane >> 4) << 3)));
  f.q[0] = *(const uint4e*)p;
  f.q[1] = *(const uint4e*)(p + 32);
  return f.v;
}

// ---------------------------------------------------------------------------
// Generic WMMA GEMM: C[M,Nout] = act( A[M,K] @ W[Nout,K]^T + bias )
// wave = 64 rows x 16 cols; block = 8 waves = 64 rows x 128 cols
// AUX: synthesize last K-chunk (cols 512=ctx, 513=y, rest zero) in registers.
// ---------------------------------------------------------------------------
enum { AUX_NONE = 0, AUX_ROW = 1, AUX_CONST = 2 };

template <int AUX, bool RELU, bool OUTBF>
__launch_bounds__(256, 1)
__global__ void gemm_wmma_k(const __bf16* __restrict__ A, int lda,
                            const __bf16* __restrict__ W, int ldw,
                            const float* __restrict__ bias,
                            const float* __restrict__ auxc,
                            const float* __restrict__ auxy,
                            void* __restrict__ Cout, int ldc, int ktiles) {
  const int lane = threadIdx.x & 31;
  const int wave = threadIdx.x >> 5;
  const int row0 = blockIdx.x * 64;
  const int col0 = blockIdx.y * 128 + wave * 16;

  v8f acc[4];
  acc[0] = (v8f)0.f; acc[1] = (v8f)0.f; acc[2] = (v8f)0.f; acc[3] = (v8f)0.f;

  for (int kt = 0; kt < ktiles; ++kt) {
    const int k0 = kt * 32;
    v16bf b = load_b(W, ldw, col0 + (lane & 15), k0, lane);
#pragma unroll
    for (int s = 0; s < 4; ++s) {
      const int m = row0 + s * 16 + (lane & 15);
      v16bf a;
      if (AUX != AUX_NONE && kt == ktiles - 1) {
        FragBF f;
        f.q[0] = (uint4e)0u; f.q[1] = (uint4e)0u;
        if (lane < 16) {   // k=512 -> ctx, k=513 -> y, rest zero
          float cv, yv;
          if (AUX == AUX_ROW) { cv = auxc[m]; yv = auxy[m]; }
          else               { cv = auxc[0]; yv = auxy[0]; }
          f.v[0] = (__bf16)cv;
          f.v[1] = (__bf16)yv;
        }
        a = f.v;
      } else {
        a = load_a(A, lda, m, k0, lane);
      }
      acc[s] = __builtin_amdgcn_wmma_f32_16x16x32_bf16(
          false, a, false, b, (short)0, acc[s], false, false);
    }
  }

  const int n = col0 + (lane & 15);
  const float bv = bias ? bias[n] : 0.f;
#pragma unroll
  for (int s = 0; s < 4; ++s) {
#pragma unroll
    for (int r = 0; r < 8; ++r) {
      const int m = row0 + s * 16 + ((lane >> 4) << 3) + r;
      float v = acc[s][r] + bv;
      if (RELU) v = fmaxf(v, 0.f);
      if (OUTBF) ((__bf16*)Cout)[(size_t)m * ldc + n] = (__bf16)v;
      else       ((float*)Cout)[(size_t)m * ldc + n] = v;
    }
  }
}

// ---------------------------------------------------------------------------
// Fused final step: gates = xN @ Wih^T + gbias (gbias = h@Whh^T + bih + bhh),
// LSTM elementwise in-register, emit h_new (bf16).
// Tile: wave = 32 rows x 16 hidden cols x 4 gates = 8 accumulators (64 VGPRs);
// only one B fragment live at a time -> no scratch spills around WMMA.
// ---------------------------------------------------------------------------
__launch_bounds__(256, 1)
__global__ void gemm_lstm_final_k(const __bf16* __restrict__ xN,
                                  const __bf16* __restrict__ Wih,
                                  const float* __restrict__ gbias,
                                  const float* __restrict__ cbuf,
                                  __bf16* __restrict__ hN) {
  const int lane = threadIdx.x & 31;
  const int wave = threadIdx.x >> 5;
  const int row0 = blockIdx.x * 32;
  const int col0 = blockIdx.y * 128 + wave * 16;

  v8f acc[2][4];
#pragma unroll
  for (int s = 0; s < 2; ++s)
#pragma unroll
    for (int g = 0; g < 4; ++g) acc[s][g] = (v8f)0.f;

  for (int kt = 0; kt < 32; ++kt) {
    const int k0 = kt * 32;
    v16bf a0 = load_a(xN, 1024, row0 + (lane & 15), k0, lane);
    v16bf a1 = load_a(xN, 1024, row0 + 16 + (lane & 15), k0, lane);
#pragma unroll
    for (int g = 0; g < 4; ++g) {
      v16bf b = load_b(Wih, 1024, g * 1024 + col0 + (lane & 15), k0, lane);
      acc[0][g] = __builtin_amdgcn_wmma_f32_16x16x32_bf16(
          false, a0, false, b, (short)0, acc[0][g], false, false);
      acc[1][g] = __builtin_amdgcn_wmma_f32_16x16x32_bf16(
          false, a1, false, b, (short)0, acc[1][g], false, false);
    }
  }

  const int n = col0 + (lane & 15);
  const float cprev = cbuf[n];
  const float gbi = gbias[n];
  const float gbf = gbias[1024 + n];
  const float gbg = gbias[2048 + n];
  const float gbo = gbias[3072 + n];
#pragma unroll
  for (int s = 0; s < 2; ++s) {
#pragma unroll
    for (int r = 0; r < 8; ++r) {
      const int m = row0 + s * 16 + ((lane >> 4) << 3) + r;
      const float gi = acc[s][0][r] + gbi;
      const float gf = acc[s][1][r] + gbf;
      const float gg = acc[s][2][r] + gbg;
      const float go = acc[s][3][r] + gbo;
      const float cn = sigm(gf) * cprev + sigm(gi) * tanhf(gg);
      const float hn = sigm(go) * tanhf(cn);
      hN[(size_t)m * 1024 + n] = (__bf16)hn;
    }
  }
}

// ---------------------------------------------------------------------------
// Sequential LSTM scan: 32 persistent WGs x 256 threads. Each WG owns hidden
// slice j in [wg*32, wg*32+32) -> 128 rows of Whh (4 gates) resident in LDS
// (256 KB of the 320 KB/WGP). c kept in registers of lanes 0..31 of wave 0.
// One monotonic global-atomic grid barrier per timestep.
// ---------------------------------------------------------------------------
__device__ __forceinline__ float dot512_lds(const __bf16* w, const float* h) {
  const uint4e* wq = (const uint4e*)w;
  float s = 0.f;
#pragma unroll 4
  for (int i = 0; i < 64; ++i) {
    uint4e q = wq[i];               // 8 bf16 via ds b128
    const float* hp = h + i * 8;
    s += bflo(q.x) * hp[0] + bfhi(q.x) * hp[1];
    s += bflo(q.y) * hp[2] + bfhi(q.y) * hp[3];
    s += bflo(q.z) * hp[4] + bfhi(q.z) * hp[5];
    s += bflo(q.w) * hp[6] + bfhi(q.w) * hp[7];
  }
  return s;
}

__launch_bounds__(256, 1)
__global__ void lstm_seq_k(const __bf16* __restrict__ Whh,
                           const float* __restrict__ xWih,   // [T,4096] f32
                           const float* __restrict__ bih,
                           const float* __restrict__ bhh,
                           float* __restrict__ hb0, float* __restrict__ hb1,
                           float* __restrict__ cbuf, float* __restrict__ gbias,
                           unsigned* __restrict__ bar) {
  extern __shared__ char smem[];
  __bf16* wsl = (__bf16*)smem;                          // 128*1024 bf16 = 256 KB
  float*  hs  = (float*)(smem + 128 * 1024 * 2);        // 1024 f32
  float*  sg  = (float*)(smem + 128 * 1024 * 2 + 4096); // 128 f32

  const int tid  = threadIdx.x;
  const int wg   = blockIdx.x;          // 0..31
  const int r    = tid >> 1;            // local gate-row 0..127
  const int half = tid & 1;
  const int grow = ((r >> 5) << 10) + wg * 32 + (r & 31); // gate*1024 + j

  { // stage this WG's Whh slice into LDS (each thread: 512 bf16 = 64 x b128)
    const uint4e* src = (const uint4e*)(Whh + (size_t)grow * 1024 + half * 512);
    uint4e* dst = (uint4e*)(wsl + r * 1024 + half * 512);
    for (int i = 0; i < 64; ++i) dst[i] = src[i];
  }
  float creg = 0.f;
  __syncthreads();

  for (int t = 0; t < 2048; ++t) {
    float* hcur  = (t & 1) ? hb1 : hb0;
    float* hnext = (t & 1) ? hb0 : hb1;

    ((flt4e*)hs)[tid] = ((const flt4e*)hcur)[tid];  // stage h (4 KB)
    __syncthreads();

    float sum = dot512_lds(wsl + r * 1024 + half * 512, hs + half * 512);
    sum += __shfl_xor(sum, 1);
    if (half == 0)
      sg[r] = sum + xWih[(size_t)t * 4096 + grow] + bih[grow] + bhh[grow];
    __syncthreads();

    if (tid < 32) {
      const float gi = sg[tid], gf = sg[32 + tid], gg = sg[64 + tid], go = sg[96 + tid];
      creg = sigm(gf) * creg + sigm(gi) * tanhf(gg);
      hnext[wg * 32 + tid] = sigm(go) * tanhf(creg);
    }

    // ---- grid barrier (monotonic counter; re-zeroed by init each launch) ----
    __threadfence();
    __syncthreads();
    if (tid == 0) {
      atomicAdd(bar, 1u);
      const unsigned target = 32u * (unsigned)(t + 1);
      while (__hip_atomic_load(bar, __ATOMIC_RELAXED, __HIP_MEMORY_SCOPE_AGENT) < target)
        __builtin_amdgcn_s_sleep(1);
    }
    __syncthreads();
    __threadfence();
  }

  // Final: gbias = h_final @ Whh^T + bih + bhh   (h_final is in hb0: 2048 even)
  ((flt4e*)hs)[tid] = ((const flt4e*)hb0)[tid];
  __syncthreads();
  float sum = dot512_lds(wsl + r * 1024 + half * 512, hs + half * 512);
  sum += __shfl_xor(sum, 1);
  if (half == 0) gbias[grow] = sum + bih[grow] + bhh[grow];
  if (tid < 32) cbuf[wg * 32 + tid] = creg;
}

// ---------------------------------------------------------------------------
// Small helpers
// ---------------------------------------------------------------------------
__global__ void f2bf_k(const float* __restrict__ src, __bf16* __restrict__ dst, int n) {
  const int i = blockIdx.x * 256 + threadIdx.x;
  if (i < n) dst[i] = (__bf16)src[i];
}

__global__ void padW1_k(const float* __restrict__ W1, __bf16* __restrict__ W1p) {
  const int i = blockIdx.x * 256 + threadIdx.x;     // over 1024*544
  if (i < 1024 * 544) {
    const int rr = i / 544, c = i - rr * 544;
    W1p[i] = (__bf16)(c < 514 ? W1[rr * 514 + c] : 0.f);
  }
}

__global__ void init0_k(float* __restrict__ hb0, unsigned* __restrict__ bar) {
  const int i = threadIdx.x;
  hb0[i] = 0.f;
  if (i == 0) *bar = 0u;
}

// out[m] = dot(h_new[m,:], Wout)  — deterministic tree reduction
__global__ void rowdot_k(const __bf16* __restrict__ hN, const float* __restrict__ Wout,
                         float* __restrict__ out) {
  const int m = blockIdx.x, tid = threadIdx.x;
  float s = 0.f;
  for (int i = tid; i < 1024; i += 256) s += (float)hN[(size_t)m * 1024 + i] * Wout[i];
  __shared__ float red[256];
  red[tid] = s;
  __syncthreads();
  for (int off = 128; off > 0; off >>= 1) {
    if (tid < off) red[tid] += red[tid + off];
    __syncthreads();
  }
  if (tid == 0) out[m] = red[0];
}

// ---------------------------------------------------------------------------
// Launcher
// ---------------------------------------------------------------------------
extern "C" void kernel_launch(void* const* d_in, const int* in_sizes, int n_in,
                              void* d_out, int out_size, void* d_ws, size_t ws_size,
                              hipStream_t stream) {
  (void)in_sizes; (void)n_in; (void)out_size; (void)ws_size;
  constexpr int N = 16384, T = 2048, F = 512, H = 1024;

  const float* X        = (const float*)d_in[0];
  const float* hist_X   = (const float*)d_in[1];
  const float* hist_y   = (const float*)d_in[2];
  const float* hist_ctx = (const float*)d_in[3];
  const float* W1       = (const float*)d_in[4];
  const float* b1       = (const float*)d_in[5];
  const float* W2       = (const float*)d_in[6];
  const float* b2       = (const float*)d_in[7];
  const float* Wih      = (const float*)d_in[8];
  const float* Whh      = (const float*)d_in[9];
  const float* bih      = (const float*)d_in[10];
  const float* bhh      = (const float*)d_in[11];
  const float* Wout     = (const float*)d_in[12];

  char* ws = (char*)d_ws;
  size_t off = 0;
  auto alloc = [&](size_t bytes) -> void* {
    void* p = ws + off;
    off += (bytes + 255) & ~(size_t)255;
    return p;
  };
  __bf16* Xbf   = (__bf16*)alloc((size_t)N * F * 2);        // 16 MB
  __bf16* HXbf  = (__bf16*)alloc((size_t)T * F * 2);        // 2 MB
  __bf16* W1p   = (__bf16*)alloc((size_t)H * 544 * 2);      // 1.1 MB
  __bf16* W2b   = (__bf16*)alloc((size_t)H * H * 2);        // 2 MB
  __bf16* Wihb  = (__bf16*)alloc((size_t)4 * H * H * 2);    // 8 MB
  __bf16* Whhb  = (__bf16*)alloc((size_t)4 * H * H * 2);    // 8 MB
  __bf16* h1h   = (__bf16*)alloc((size_t)T * H * 2);        // 4 MB
  __bf16* xh    = (__bf16*)alloc((size_t)T * H * 2);        // 4 MB
  float*  xwih  = (float*)alloc((size_t)T * 4 * H * 4);     // 32 MB
  __bf16* h1N   = (__bf16*)alloc((size_t)N * H * 2);        // 32 MB (reused as hN)
  __bf16* xN    = (__bf16*)alloc((size_t)N * H * 2);        // 32 MB
  float*  hb0   = (float*)alloc(H * 4);
  float*  hb1   = (float*)alloc(H * 4);
  float*  cbuf  = (float*)alloc(H * 4);
  float*  gbias = (float*)alloc(4 * H * 4);
  unsigned* bar = (unsigned*)alloc(256);
  __bf16* hN = h1N;  // h1N dead after N fe2 -> reuse for h_new

  // --- precision conversion + padding + init ---
  f2bf_k<<<(N * F + 255) / 256, 256, 0, stream>>>(X, Xbf, N * F);
  f2bf_k<<<(T * F + 255) / 256, 256, 0, stream>>>(hist_X, HXbf, T * F);
  f2bf_k<<<(H * H + 255) / 256, 256, 0, stream>>>(W2, W2b, H * H);
  f2bf_k<<<(4 * H * H + 255) / 256, 256, 0, stream>>>(Wih, Wihb, 4 * H * H);
  f2bf_k<<<(4 * H * H + 255) / 256, 256, 0, stream>>>(Whh, Whhb, 4 * H * H);
  padW1_k<<<(1024 * 544 + 255) / 256, 256, 0, stream>>>(W1, W1p);
  init0_k<<<1, 1024, 0, stream>>>(hb0, bar);

  // --- history feature extractor + xWih, batched over all T (WMMA) ---
  gemm_wmma_k<AUX_ROW, true, true><<<dim3(T / 64, H / 128), 256, 0, stream>>>(
      HXbf, F, W1p, 544, b1, hist_ctx, hist_y, h1h, H, 17);
  gemm_wmma_k<AUX_NONE, true, true><<<dim3(T / 64, H / 128), 256, 0, stream>>>(
      h1h, H, W2b, H, b2, nullptr, nullptr, xh, H, 32);
  gemm_wmma_k<AUX_NONE, false, false><<<dim3(T / 64, (4 * H) / 128), 256, 0, stream>>>(
      xh, H, Wihb, H, nullptr, nullptr, nullptr, xwih, 4 * H, 32);

  // --- candidate feature extractor, batched over N (WMMA) ---
  gemm_wmma_k<AUX_CONST, true, true><<<dim3(N / 64, H / 128), 256, 0, stream>>>(
      Xbf, F, W1p, 544, b1, hist_ctx + (T - 1), hist_y + (T - 1), h1N, H, 17);
  gemm_wmma_k<AUX_NONE, true, true><<<dim3(N / 64, H / 128), 256, 0, stream>>>(
      h1N, H, W2b, H, b2, nullptr, nullptr, xN, H, 32);

  // --- sequential recurrence: Whh LDS-resident, 1 grid barrier / step ---
  hipFuncSetAttribute((const void*)lstm_seq_k,
                      hipFuncAttributeMaxDynamicSharedMemorySize, 266752);
  lstm_seq_k<<<32, 256, 266752, stream>>>(Whhb, xwih, bih, bhh,
                                          hb0, hb1, cbuf, gbias, bar);

  // --- fused final gates GEMM + LSTM elementwise (WMMA) ---
  gemm_lstm_final_k<<<dim3(N / 32, H / 128), 256, 0, stream>>>(
      xN, Wihb, gbias, cbuf, hN);

  // --- out = h_new @ Wout^T ---
  rowdot_k<<<N, 256, 0, stream>>>(hN, Wout, (float*)d_out);
}
